// CausalSelfAttention_56762287784278
// MI455X (gfx1250) — compile-verified, matrix operand order checked by reading, not run on hardware
//
#include <hip/hip_runtime.h>

// ---------------------------------------------------------------------------
// Causal self-attention for MI455X (gfx1250, wave32, WMMA 16x16x32 f16->f32).
//
// - fp32 inputs converted to f16, all GEMMs accumulate fp32 via
//   v_wmma_f32_16x16x32_f16; softmax statistics in fp32.
// - LDS staging is done by the Tensor Data Mover (tensor_load_to_lds, tracked
//   by TENSORcnt).  TDM's pad_interval/pad_amount reproduce our bank-conflict
//   padding (GEMM: 32h rows -> stride 40h; attention: 64h rows -> stride 72h).
// - GEMM is double-buffered: wave 0 launches the next K-tile's two DMAs, then
//   s_wait_tensorcnt(2) (completions are in-order) before releasing compute.
//
// Fragment layout (CDNA5 ISA 7.12.2, 16-bit A 16x32; B mirrored lane<->N so
// B operands are loaded from N-major storage with identical addressing):
//   lane m      : row m, K = {col0+0..7} in h[0], {col0+16..23} in h[1]
//   lane m+16   : row m, K = {col0+8..15} in h[0], {col0+24..31} in h[1]
// ---------------------------------------------------------------------------

typedef __attribute__((ext_vector_type(16))) _Float16 v16h;
typedef __attribute__((ext_vector_type(8)))  _Float16 v8h;
typedef __attribute__((ext_vector_type(8)))  float    v8f;
typedef __attribute__((ext_vector_type(4)))  unsigned int v4u;
typedef __attribute__((ext_vector_type(8)))  int      v8i;
typedef __attribute__((ext_vector_type(4)))  int      v4i;

#define DEV __device__ __forceinline__

static DEV v8f vzero8() {
  v8f z = {0.f, 0.f, 0.f, 0.f, 0.f, 0.f, 0.f, 0.f};
  return z;
}

static DEV v16h load_frag(const _Float16* base, int row0, int col0, int ld, int lane) {
  const _Float16* p = base + (size_t)(row0 + (lane & 15)) * ld + col0 + ((lane >> 4) * 8);
  union { v16h v; v8h h[2]; } u;
  u.h[0] = *(const v8h*)(p);
  u.h[1] = *(const v8h*)(p + 16);
  return u.v;
}

static DEV v8f wmma_f16(v16h a, v16h b, v8f c) {
  return __builtin_amdgcn_wmma_f32_16x16x32_f16(false, a, false, b, (short)0, c,
                                                false, false);
}

// LDS byte offset of a __shared__ object: generic LDS-aperture addresses
// truncate to the LDS byte address (ISA 10.2 aperture mapping).
static DEV unsigned int lds_off(const void* p) {
  return (unsigned int)(unsigned long long)p;
}

// TDM 2D f16 tile load: tile_d0 contiguous elements per row, tile_d1 rows,
// row stride in elements.  pad_code_i: LDS pad interval = 2^(code+1) DWORDs;
// pad_code_a: pad amount = (code+1) DWORDs.  D# built per ISA ch.8.
// This toolchain exposes the 6-arg tensor_load_to_lds builtin
// (g0 u32x4, g1 i32x8, g2 i32x4, g3 i32x4, i32x8, cpol).
static DEV void tdm_load_2d_f16(unsigned int lds_addr, const _Float16* gptr,
                                unsigned int tile_d0, unsigned int tile_d1,
                                unsigned long long row_stride,
                                unsigned int pad_code_i, unsigned int pad_code_a) {
  unsigned long long ga = (unsigned long long)(const void*)gptr;
  v4u g0;
  g0[0] = 1u;                                   // count=1 (valid user D#)
  g0[1] = lds_addr;                             // lds_addr[31:0]
  g0[2] = (unsigned int)ga;                     // global_addr[31:0]
  g0[3] = (unsigned int)((ga >> 32) & 0x01ffffffu) | (2u << 30);  // addr hi, type=2
  v8i g1;
  g1[0] = (int)((1u << 16) |                    // data_size = 2 bytes
                (1u << 20) |                    // pad_enable
                (pad_code_i << 22) | (pad_code_a << 25));
  g1[1] = (int)((tile_d0 & 0xffffu) << 16);     // tensor_dim0[15:0] (== tile_d0)
  g1[2] = (int)(((tile_d0 >> 16) & 0xffffu) | ((tile_d1 & 0xffffu) << 16));
  g1[3] = (int)(((tile_d1 >> 16) & 0xffffu) | ((tile_d0 & 0xffffu) << 16)); // tile_dim0
  g1[4] = (int)(tile_d1 & 0xffffu);             // tile_dim1, tile_dim2=0
  g1[5] = (int)(unsigned int)row_stride;        // tensor_dim0_stride[31:0]
  g1[6] = (int)((row_stride >> 32) & 0xffffu);  // stride[47:32]; dim1_stride=0 (2D)
  g1[7] = 0;
  v4i z4 = {0, 0, 0, 0};
  v8i z8 = {0, 0, 0, 0, 0, 0, 0, 0};
  __builtin_amdgcn_tensor_load_to_lds(g0, g1, z4, z4, z8, 0);
}

// ---------------------------------------------------------------------------
// Precision conversion / weight transposition (one-shot, bandwidth trivial).
// ---------------------------------------------------------------------------
__global__ void cvt_f32_to_f16(const float* __restrict__ in,
                               _Float16* __restrict__ out, int n) {
  int i = blockIdx.x * blockDim.x + threadIdx.x;
  int stride = gridDim.x * blockDim.x;
  for (; i < n; i += stride) out[i] = (_Float16)in[i];
}

// in: [K][N] f32 (row-major)  ->  out: [N][K] f16 (row-major, i.e. W^T)
__global__ void transpose_f32_to_f16(const float* __restrict__ in,
                                     _Float16* __restrict__ out, int K, int N) {
  int i = blockIdx.x * blockDim.x + threadIdx.x;
  int stride = gridDim.x * blockDim.x;
  int total = K * N;
  for (; i < total; i += stride) {
    int k = i % K, n = i / K;
    out[i] = (_Float16)in[(size_t)k * N + n];
  }
}

// ---------------------------------------------------------------------------
// 128x128x32-tiled GEMM, 256 threads = 8 waves (2 M x 4 N), 64x32 per wave.
// TDM-staged, double-buffered LDS.
// MODE 0: QKV projection -> scatter into per-head Q[t][d], K[t][d], V^T[d][t]
// MODE 1: output projection -> fp32 d_out + bias
// ---------------------------------------------------------------------------
template <int MODE>
__global__ __launch_bounds__(256) void gemm128(
    const _Float16* __restrict__ A,   // [M][K] row-major
    const _Float16* __restrict__ Bt,  // [N][K] row-major (W^T)
    const float* __restrict__ bias,   // [N]
    int K,
    _Float16* __restrict__ qh, _Float16* __restrict__ kh,
    _Float16* __restrict__ vTh, float* __restrict__ outf) {
  constexpr int BM = 128, BN = 128, BK = 32, LD = BK + 8;  // TDM-padded 80B rows
  __shared__ alignas(16) _Float16 As[2][BM * LD];
  __shared__ alignas(16) _Float16 Bs[2][BM * LD];

  const int tid = threadIdx.x;
  const int lane = tid & 31, wave = tid >> 5;
  const int wmi = wave >> 2, wni = wave & 3;  // 2 x 4 wave grid
  const int m0 = blockIdx.y * BM, n0 = blockIdx.x * BN;
  const int nK = K / BK;

  v8f acc[4][2];
  for (int i = 0; i < 4; ++i)
    for (int j = 0; j < 2; ++j) acc[i][j] = vzero8();

  if (wave == 0) {  // prologue: DMA first K-tile into buffer 0
    tdm_load_2d_f16(lds_off(&As[0][0]), &A[(size_t)m0 * K], BK, BM, K, 3, 3);
    tdm_load_2d_f16(lds_off(&Bs[0][0]), &Bt[(size_t)n0 * K], BK, BM, K, 3, 3);
  }

  for (int kt = 0; kt < nK; ++kt) {
    const int cur = kt & 1;
    __syncthreads();  // readers of buf[cur^1] (iteration kt-1) are done
    if (wave == 0) {
      if (kt + 1 < nK) {  // DMA next tile while this one is consumed
        tdm_load_2d_f16(lds_off(&As[cur ^ 1][0]),
                        &A[(size_t)m0 * K + (size_t)(kt + 1) * BK], BK, BM, K, 3, 3);
        tdm_load_2d_f16(lds_off(&Bs[cur ^ 1][0]),
                        &Bt[(size_t)n0 * K + (size_t)(kt + 1) * BK], BK, BM, K, 3, 3);
        __builtin_amdgcn_s_wait_tensorcnt(2);  // in-order: oldest 2 (cur) done
      } else {
        __builtin_amdgcn_s_wait_tensorcnt(0);
      }
    }
    __syncthreads();  // buf[cur] ready for everyone

    v16h a[4], b[2];
    for (int mt = 0; mt < 4; ++mt)
      a[mt] = load_frag(As[cur], wmi * 64 + mt * 16, 0, LD, lane);
    for (int nt = 0; nt < 2; ++nt)
      b[nt] = load_frag(Bs[cur], wni * 32 + nt * 16, 0, LD, lane);
    for (int mt = 0; mt < 4; ++mt)
      for (int nt = 0; nt < 2; ++nt)
        acc[mt][nt] = wmma_f16(a[mt], b[nt], acc[mt][nt]);
  }

  const int hi8 = (lane & 16) ? 8 : 0;
  const int cl = lane & 15;
  for (int mt = 0; mt < 4; ++mt)
    for (int nt = 0; nt < 2; ++nt)
      for (int r = 0; r < 8; ++r) {
        int rowg = m0 + wmi * 64 + mt * 16 + r + hi8;  // token index b*2048+t
        int colg = n0 + wni * 32 + nt * 16 + cl;       // column in [0, N)
        float v = acc[mt][nt][r] + bias[colg];
        if (MODE == 0) {
          int which = colg >> 10, rem = colg & 1023;
          int hh = rem >> 6, dd = rem & 63;
          int bb = rowg >> 11, tt = rowg & 2047;
          size_t bh = (size_t)bb * 16 + hh;
          if (which == 0)      qh[(bh * 2048 + tt) * 64 + dd]  = (_Float16)v;
          else if (which == 1) kh[(bh * 2048 + tt) * 64 + dd]  = (_Float16)v;
          else                 vTh[(bh * 64 + dd) * 2048 + tt] = (_Float16)v;  // V^T
        } else {
          outf[(size_t)rowg * 1024 + colg] = v;
        }
      }
}

// ---------------------------------------------------------------------------
// Flash attention: one block = one (b,h) x 64-query tile. 256 threads / 8
// waves; wave w owns S/O row-tile w>>1 and column-tile pair (w&1)*2+{0,1}.
// Q/K/V^T tiles staged by TDM.
// ---------------------------------------------------------------------------
__global__ __launch_bounds__(256) void attn64(
    const _Float16* __restrict__ qh, const _Float16* __restrict__ kh,
    const _Float16* __restrict__ vTh, _Float16* __restrict__ ao) {
  constexpr int LQ = 72, LS = 66;  // TDM-padded 144B rows / 264B f32 rows
  __shared__ alignas(16) _Float16 Qs[64 * LQ];
  __shared__ alignas(16) _Float16 Ks[64 * LQ];
  __shared__ alignas(16) _Float16 VTs[64 * LQ];  // [d][t']
  __shared__ alignas(16) _Float16 Ps[64 * LQ];
  __shared__ float Ss[64 * LS];
  __shared__ float m_s[64], l_s[64], al_s[64];

  const int tid = threadIdx.x, lane = tid & 31, wave = tid >> 5;
  const int bh = blockIdx.x >> 5;  // b*16 + h
  const int qb = blockIdx.x & 31;  // query block (64 rows each)
  const int b = bh >> 4, h = bh & 15;

  const _Float16* qp = qh + (size_t)bh * 2048 * 64;
  const _Float16* kp = kh + (size_t)bh * 2048 * 64;
  const _Float16* vp = vTh + (size_t)bh * 64 * 2048;

  if (wave == 0) {  // DMA 64x64 Q tile, padded to stride 72h
    tdm_load_2d_f16(lds_off(&Qs[0]), &qp[(size_t)(qb * 64) * 64], 64, 64, 64, 4, 3);
    __builtin_amdgcn_s_wait_tensorcnt(0);
  }
  if (tid < 64) { m_s[tid] = -1e30f; l_s[tid] = 0.f; }
  __syncthreads();

  const int rt = wave >> 1;       // row tile (queries)
  const int pb = (wave & 1) * 2;  // key-tile / O-col-tile pair base
  const int hi8 = (lane & 16) ? 8 : 0;
  const int cl = lane & 15;

  v16h qf[2];
  qf[0] = load_frag(Qs, rt * 16, 0, LQ, lane);
  qf[1] = load_frag(Qs, rt * 16, 32, LQ, lane);

  v8f o[2];
  o[0] = vzero8(); o[1] = vzero8();

  for (int kb = 0; kb <= qb; ++kb) {
    // End-of-iteration barrier guarantees all readers of Ks/VTs/Ps are done.
    if (wave == 0) {
      tdm_load_2d_f16(lds_off(&Ks[0]), &kp[(size_t)(kb * 64) * 64], 64, 64, 64, 4, 3);
      tdm_load_2d_f16(lds_off(&VTs[0]), &vp[(size_t)(kb * 64)], 64, 64, 2048, 4, 3);
      __builtin_amdgcn_s_wait_tensorcnt(0);
    }
    __syncthreads();

    // S = Q K^T * 1/sqrt(64), causal mask on the diagonal block.
    for (int kt2 = 0; kt2 < 2; ++kt2) {
      int kt = pb + kt2;
      v8f s = vzero8();
      s = wmma_f16(qf[0], load_frag(Ks, kt * 16, 0, LQ, lane), s);
      s = wmma_f16(qf[1], load_frag(Ks, kt * 16, 32, LQ, lane), s);
      for (int r = 0; r < 8; ++r) {
        int rowl = rt * 16 + r + hi8;
        int coll = kt * 16 + cl;
        float v = s[r] * 0.125f;
        if (kb == qb && coll > rowl) v = -1e9f;
        Ss[rowl * LS + coll] = v;
      }
    }
    __syncthreads();

    // Online softmax: 4 lanes per row, reduce via wave32 shuffles.
    {
      int row = tid >> 2, quad = tid & 3;
      float m_old = m_s[row];
      const float* srow = &Ss[row * LS + quad * 16];
      float mx = srow[0];
      for (int j = 1; j < 16; ++j) mx = fmaxf(mx, srow[j]);
      mx = fmaxf(mx, __shfl_xor(mx, 1, 32));
      mx = fmaxf(mx, __shfl_xor(mx, 2, 32));
      float m_new = fmaxf(m_old, mx);
      float sum = 0.f;
      _Float16* prow = &Ps[row * LQ + quad * 16];
      for (int j = 0; j < 16; ++j) {
        float p = __expf(srow[j] - m_new);
        prow[j] = (_Float16)p;
        sum += p;
      }
      sum += __shfl_xor(sum, 1, 32);
      sum += __shfl_xor(sum, 2, 32);
      if (quad == 0) {
        float alpha = __expf(m_old - m_new);
        al_s[row] = alpha;
        l_s[row] = l_s[row] * alpha + sum;
        m_s[row] = m_new;
      }
    }
    __syncthreads();

    // O = O*alpha + P V   (B operand = contiguous rows of staged V^T)
    for (int ci = 0; ci < 2; ++ci)
      for (int r = 0; r < 8; ++r) o[ci][r] *= al_s[rt * 16 + r + hi8];
    for (int kk = 0; kk < 2; ++kk) {
      v16h pf = load_frag(Ps, rt * 16, kk * 32, LQ, lane);
      for (int ci = 0; ci < 2; ++ci) {
        v16h vf = load_frag(VTs, (pb + ci) * 16, kk * 32, LQ, lane);
        o[ci] = wmma_f16(pf, vf, o[ci]);
      }
    }
    __syncthreads();
  }

  // Finalize and store f16 attention output in [B*T][1024] layout.
  for (int ci = 0; ci < 2; ++ci)
    for (int r = 0; r < 8; ++r) {
      int rowl = rt * 16 + r + hi8;
      float v = o[ci][r] / l_s[rowl];
      int qg = qb * 64 + rowl;
      int cg = (pb + ci) * 16 + cl;
      ao[((size_t)(b * 2048 + qg)) * 1024 + h * 64 + cg] = (_Float16)v;
    }
}

// ---------------------------------------------------------------------------
extern "C" void kernel_launch(void* const* d_in, const int* in_sizes, int n_in,
                              void* d_out, int out_size, void* d_ws,
                              size_t ws_size, hipStream_t stream) {
  const float* x    = (const float*)d_in[0];
  const float* Wqkv = (const float*)d_in[1];
  const float* bqkv = (const float*)d_in[2];
  const float* Wout = (const float*)d_in[3];
  const float* bout = (const float*)d_in[4];
  float* out = (float*)d_out;

  constexpr int B = 2, T = 2048, D = 1024, BT = B * T;

  char* w = (char*)d_ws;
  size_t off = 0;
  _Float16* xh    = (_Float16*)(w + off); off += (size_t)BT * D * 2;
  _Float16* WqkvT = (_Float16*)(w + off); off += (size_t)3 * D * D * 2;
  _Float16* WoutT = (_Float16*)(w + off); off += (size_t)D * D * 2;
  _Float16* qh    = (_Float16*)(w + off); off += (size_t)BT * D * 2;
  _Float16* kh    = (_Float16*)(w + off); off += (size_t)BT * D * 2;
  _Float16* vTh   = (_Float16*)(w + off); off += (size_t)BT * D * 2;
  _Float16* ao    = (_Float16*)(w + off); off += (size_t)BT * D * 2;

  cvt_f32_to_f16<<<2048, 256, 0, stream>>>(x, xh, BT * D);
  transpose_f32_to_f16<<<4096, 256, 0, stream>>>(Wqkv, WqkvT, D, 3 * D);
  transpose_f32_to_f16<<<2048, 256, 0, stream>>>(Wout, WoutT, D, D);

  gemm128<0><<<dim3(3 * D / 128, BT / 128), 256, 0, stream>>>(
      xh, WqkvT, bqkv, D, qh, kh, vTh, nullptr);

  attn64<<<B * 16 * (T / 64), 256, 0, stream>>>(qh, kh, vTh, ao);

  gemm128<1><<<dim3(D / 128, BT / 128), 256, 0, stream>>>(
      ao, WoutT, bout, D, nullptr, nullptr, nullptr, out);
}
